// Manifold_33122787787616
// MI455X (gfx1250) — matrix operand order repaired and checked by reading, not compiled, not run on hardware
//
#include <hip/hip_runtime.h>
#include <hip/hip_bf16.h>

#define B_  16
#define T_  512
#define V_  8000
#define D_  512
#define L_  4
#define H_  8
#define R_  32
#define DH_ 64
#define DT_ 0.1f
#define EPS_ 1e-5f

#define WP_ 8192   // padded N stride for packed ro_w copy (zero-filled pad)

#define BM 128
#define BN 256
#define BK 32
#define ASTRIDE 20  // ldsA row stride in u32 (16B-aligned chunks, conflict-free)

typedef __attribute__((ext_vector_type(16))) _Float16 v16h;
typedef __attribute__((ext_vector_type(8)))  float    v8f;

union FragU { unsigned u[8]; v16h v; };
union HPack { _Float16 h[2]; unsigned u; };

// ---------------------------------------------------------------------------
// Kernel 1: fp32 -> f16 conversion of ro_w into k-paired u32 words:
//   Whp[k/2][n] = pack(h(ro_w[k][n]), h(ro_w[k+1][n])), padded to WP_ cols.
// This is exactly the LDS layout the WMMA B-fragments want, so the GEMM can
// stage B tiles with straight async copies (no in-kernel repacking).
// ---------------------------------------------------------------------------
__global__ void cvt_pad_pack(const float* __restrict__ src,
                             unsigned* __restrict__ dst) {
  int i = blockIdx.x * blockDim.x + threadIdx.x;  // over (D_/2)*WP_
  int pg = i >> 13;        // / WP_  -> k-pair row 0..255
  int n  = i & (WP_ - 1);
  HPack hp;
  if (n < V_) {
    hp.h[0] = (_Float16)src[(size_t)(2 * pg) * V_ + n];
    hp.h[1] = (_Float16)src[(size_t)(2 * pg + 1) * V_ + n];
  } else {
    hp.h[0] = (_Float16)0.f;
    hp.h[1] = (_Float16)0.f;
  }
  dst[i] = hp.u;
}

// ---------------------------------------------------------------------------
// Kernel 2: sequential scan. One workgroup per batch element (latency-bound
// recurrence; batches independent -> 16 WGPs in parallel, no cross-WG sync).
// Each thread owns the adjacent pair (2*tid, 2*tid+1) so the hot loops use
// b64 (float2) global loads. State lives in LDS.
// ---------------------------------------------------------------------------
__global__ __launch_bounds__(256) void scan_kernel(
    const int* __restrict__ ids, const float* __restrict__ mask,
    const float* __restrict__ emb, const float* __restrict__ x0,
    const float* __restrict__ v0, const float* __restrict__ U,
    const float* __restrict__ Wc, const float* __restrict__ G,
    const float* __restrict__ Cw, const float* __restrict__ Cb,
    const float* __restrict__ lnw, const float* __restrict__ lnb,
    _Float16* __restrict__ Aout, float* __restrict__ out_tail) {
  __shared__ float xs[D_], vs[D_], fs[D_], ctxs[D_], fes[D_];
  __shared__ float a1s[D_], xes[D_], ves[D_];
  __shared__ float coefs[H_ * R_];
  __shared__ float redA[256], redB[256];

  const int tid = threadIdx.x;
  const int b   = blockIdx.x;
  const int d0  = tid * 2, d1 = tid * 2 + 1;  // adjacent pair
  const int h   = tid >> 5;   // head for coef phase
  const int r   = tid & 31;   // rank index for coef phase
  const int hh  = d0 >> 6;    // head for gamma/output phase (pair shares head)
  const int dh  = d0 & 63;    // even in-head offset

  {
    const float2 xi = *(const float2*)(x0 + d0);
    const float2 vi = *(const float2*)(v0 + d0);
    xs[d0] = xi.x; xs[d1] = xi.y;
    vs[d0] = vi.x; vs[d1] = vi.y;
  }
  __syncthreads();

  for (int t = 0; t < T_; ++t) {
    const int   tok = ids[b * T_ + t];
    const float mk  = mask[b * T_ + t];
    {
      const float2 ef = *(const float2*)(emb + (size_t)tok * D_ + d0);
      fs[d0] = ef.x * mk;
      fs[d1] = ef.y * mk;
    }
    ctxs[d0] = 0.f; ctxs[d1] = 0.f;
    __syncthreads();

    for (int l = 0; l < L_; ++l) {
      fes[d0] = fs[d0] + ctxs[d0];
      fes[d1] = fs[d1] + ctxs[d1];

      // ---- accel #1, phase 1: coef[h,r] from (x, v) ----
      {
        const float* xv = xs + h * DH_;
        const float* vv = vs + h * DH_;
        const float* Ub = U + ((size_t)(l * H_ + h) * DH_) * R_ + r;
        const float* Gb = G + ((size_t)(l * H_ + h) * DH_) * R_ + r;
        float s1 = 0.f, s2 = 0.f;
#pragma unroll 8
        for (int dd = 0; dd < DH_; ++dd) {
          s1 += vv[dd] * Ub[dd * R_];
          s2 += xv[dd] * Gb[dd * R_];
        }
        coefs[tid] = s1 * s1 * tanhf(s2);
      }
      __syncthreads();

      // ---- accel #1, phase 2: a1 = feff - coef @ Wc; predictor step ----
      {
        const float* Wb = Wc + ((size_t)(l * H_ + hh) * R_) * DH_ + dh;
        float gx = 0.f, gy = 0.f;
#pragma unroll 8
        for (int rr = 0; rr < R_; ++rr) {
          const float2 w = *(const float2*)(Wb + rr * DH_);
          const float  c = coefs[hh * R_ + rr];
          gx += c * w.x;
          gy += c * w.y;
        }
        a1s[d0] = fes[d0] - gx;
        a1s[d1] = fes[d1] - gy;
      }
      xes[d0] = xs[d0] + DT_ * vs[d0]; xes[d1] = xs[d1] + DT_ * vs[d1];
      ves[d0] = vs[d0] + DT_ * a1s[d0]; ves[d1] = vs[d1] + DT_ * a1s[d1];
      __syncthreads();

      // ---- accel #2, phase 1: coef from (xe, ve) ----
      {
        const float* xv = xes + h * DH_;
        const float* vv = ves + h * DH_;
        const float* Ub = U + ((size_t)(l * H_ + h) * DH_) * R_ + r;
        const float* Gb = G + ((size_t)(l * H_ + h) * DH_) * R_ + r;
        float s1 = 0.f, s2 = 0.f;
#pragma unroll 8
        for (int dd = 0; dd < DH_; ++dd) {
          s1 += vv[dd] * Ub[dd * R_];
          s2 += xv[dd] * Gb[dd * R_];
        }
        coefs[tid] = s1 * s1 * tanhf(s2);
      }
      __syncthreads();

      // ---- accel #2, phase 2 + Heun corrector update ----
      {
        const float* Wb = Wc + ((size_t)(l * H_ + hh) * R_) * DH_ + dh;
        float gx = 0.f, gy = 0.f;
#pragma unroll 8
        for (int rr = 0; rr < R_; ++rr) {
          const float2 w = *(const float2*)(Wb + rr * DH_);
          const float  c = coefs[hh * R_ + rr];
          gx += c * w.x;
          gy += c * w.y;
        }
        const float a2x = fes[d0] - gx;
        const float a2y = fes[d1] - gy;
        xs[d0] = xs[d0] + 0.5f * DT_ * (vs[d0] + ves[d0]);
        xs[d1] = xs[d1] + 0.5f * DT_ * (vs[d1] + ves[d1]);
        vs[d0] = vs[d0] + 0.5f * DT_ * (a1s[d0] + a2x);
        vs[d1] = vs[d1] + 0.5f * DT_ * (a1s[d1] + a2y);
      }
      __syncthreads();

      // ---- ctx = tanh(x @ Cw[l] + Cb[l])  (matvec, b64 weight loads) ----
      {
        const float* Cwl = Cw + (size_t)l * D_ * D_ + d0;
        float acc0 = Cb[l * D_ + d0];
        float acc1 = Cb[l * D_ + d1];
#pragma unroll 4
        for (int k = 0; k < D_; ++k) {
          const float2 w  = *(const float2*)(Cwl + (size_t)k * D_);
          const float  xk = xs[k];
          acc0 += xk * w.x;
          acc1 += xk * w.y;
        }
        ctxs[d0] = tanhf(acc0);
        ctxs[d1] = tanhf(acc1);
      }
      __syncthreads();
    }  // layers

    // ---- LayerNorm + packed f16 store of GEMM A-matrix row (m = b*T+t) ----
    redA[tid] = xs[d0] + xs[d1];
    redB[tid] = xs[d0] * xs[d0] + xs[d1] * xs[d1];
    __syncthreads();
    for (int off = 128; off > 0; off >>= 1) {
      if (tid < off) { redA[tid] += redA[tid + off]; redB[tid] += redB[tid + off]; }
      __syncthreads();
    }
    const float mu   = redA[0] * (1.0f / D_);
    const float var  = redB[0] * (1.0f / D_) - mu * mu;
    const float rstd = rsqrtf(var + EPS_);
    const size_t row = (size_t)b * T_ + t;
    HPack hp;
    hp.h[0] = (_Float16)((xs[d0] - mu) * rstd * lnw[d0] + lnb[d0]);
    hp.h[1] = (_Float16)((xs[d1] - mu) * rstd * lnw[d1] + lnb[d1]);
    *(unsigned*)(Aout + row * D_ + d0) = hp.u;
    __syncthreads();
  }  // timesteps

  {
    float2 xo; xo.x = xs[d0]; xo.y = xs[d1];
    float2 vo; vo.x = vs[d0]; vo.y = vs[d1];
    *(float2*)(out_tail + b * D_ + d0) = xo;
    *(float2*)(out_tail + B_ * D_ + b * D_ + d0) = vo;
  }
}

// ---------------------------------------------------------------------------
// Kernel 3: readout GEMM, logits = A(8192x512,f16) @ Wh(512xWP_,f16) + bias.
// WMMA f32_16x16x32_f16; 128x256 block tile, 8 waves = 2(M) x 4(N),
// each wave owns a 64x64 macro-tile = 4x4 WMMA accumulators.
// Tile staging uses gfx1250 async global->LDS copies (ASYNCcnt); both
// operands are pre-laid-out so the copies are straight 16B chunks.
// ---------------------------------------------------------------------------
__global__ __launch_bounds__(256) void readout_gemm(
    const _Float16* __restrict__ A, const unsigned* __restrict__ Whp,
    const float* __restrict__ bias, float* __restrict__ Cout) {
  __shared__ unsigned ldsA[BM * ASTRIDE];   // [m][k-pair], stride 20 u32
  __shared__ unsigned ldsB[(BK / 2) * BN];  // [k-pair][n]

  const int tid     = threadIdx.x;
  const int lane    = tid & 31;
  const int wave    = tid >> 5;
  const int halfsel = lane >> 4;   // 0: lanes 0-15, 1: lanes 16-31
  const int l16     = lane & 15;
  const int waveM   = wave >> 2;   // 0..1
  const int waveN   = wave & 3;    // 0..3
  const int bm0     = blockIdx.x * BM;
  const int bn0     = blockIdx.y * BN;

  v8f acc[4][4];
#pragma unroll
  for (int i = 0; i < 4; ++i)
#pragma unroll
    for (int j = 0; j < 4; ++j)
#pragma unroll
      for (int q = 0; q < 8; ++q) acc[i][j][q] = 0.f;

  for (int k0 = 0; k0 < D_; k0 += BK) {
    __syncthreads();  // previous iteration's fragment reads are done

    // ---- stage A tile (128 rows x 32 halfs) via async copy, 2 chunks/thread
#pragma unroll
    for (int it = 0; it < 2; ++it) {
      const int id = tid + it * 256;
      const int m  = id >> 2;
      const int p0 = (id & 3) * 4;  // k-pair offset within row (16B chunks)
      const _Float16* gp = A + (size_t)(bm0 + m) * D_ + k0 + p0 * 2;
      const unsigned lo = (unsigned)(uintptr_t)&ldsA[m * ASTRIDE + p0];
      asm volatile("global_load_async_to_lds_b128 %0, %1, off"
                   :: "v"(lo), "v"((unsigned long long)(uintptr_t)gp)
                   : "memory");
    }
    // ---- stage B tile (16 k-pair rows x 256 u32) via async copy, 4/thread
    const int pg0 = k0 >> 1;
#pragma unroll
    for (int it = 0; it < 4; ++it) {
      const int id = tid + it * 256;
      const int p  = id >> 6;          // 0..15 (k-pair row)
      const int nc = (id & 63) * 4;    // 0..252 (u32 columns, 16B chunks)
      const unsigned* gp = Whp + (size_t)(pg0 + p) * WP_ + bn0 + nc;
      const unsigned lo = (unsigned)(uintptr_t)&ldsB[p * BN + nc];
      asm volatile("global_load_async_to_lds_b128 %0, %1, off"
                   :: "v"(lo), "v"((unsigned long long)(uintptr_t)gp)
                   : "memory");
    }
    // prefetch next k-slab toward this WGP (global_prefetch_b8)
    if (k0 + BK < D_) {
      __builtin_prefetch(A + (size_t)(bm0 + (tid >> 1)) * D_ + k0 + BK +
                             (tid & 1) * 16, 0, 1);
      __builtin_prefetch(Whp + (size_t)(pg0 + 16 + (tid >> 4)) * WP_ + bn0 +
                             (tid & 15) * 16, 0, 1);
    }
    asm volatile("s_wait_asynccnt 0" ::: "memory");
    __syncthreads();  // all waves' async copies are visible in LDS

    // ---- fragments in documented V_WMMA 16x16x32 f16 striping ----
    FragU afr[4];
#pragma unroll
    for (int i = 0; i < 4; ++i) {
      const int m = waveM * 64 + i * 16 + l16;
#pragma unroll
      for (int j = 0; j < 8; ++j) {
        const int p = (j < 4) ? (halfsel * 4 + j) : (8 + halfsel * 4 + (j - 4));
        afr[i].u[j] = ldsA[m * ASTRIDE + p];
      }
    }
    FragU bfr[4];
#pragma unroll
    for (int j = 0; j < 4; ++j) {
      const int n = waveN * 64 + j * 16 + l16;
#pragma unroll
      for (int q = 0; q < 8; ++q) {
        const int p = (q < 4) ? (halfsel * 4 + q) : (8 + halfsel * 4 + (q - 4));
        bfr[j].u[q] = ldsB[p * BN + n];
      }
    }

#pragma unroll
    for (int i = 0; i < 4; ++i)
#pragma unroll
      for (int j = 0; j < 4; ++j)
        acc[i][j] = __builtin_amdgcn_wmma_f32_16x16x32_f16(
            false, afr[i].v, false, bfr[j].v, (short)0, acc[i][j], false,
            false);
  }

  // epilogue: D layout -> lanes 0-15: M=r, lanes 16-31: M=r+8; N = l16
#pragma unroll
  for (int i = 0; i < 4; ++i) {
#pragma unroll
    for (int j = 0; j < 4; ++j) {
      const int gcol = bn0 + waveN * 64 + j * 16 + l16;
      if (gcol < V_) {
        const float bb = bias[gcol];
#pragma unroll
        for (int q = 0; q < 8; ++q) {
          const int grow = bm0 + waveM * 64 + i * 16 + halfsel * 8 + q;
          Cout[(size_t)grow * V_ + gcol] = acc[i][j][q] + bb;
        }
      }
    }
  }
}

// ---------------------------------------------------------------------------
extern "C" void kernel_launch(void* const* d_in, const int* in_sizes, int n_in,
                              void* d_out, int out_size, void* d_ws,
                              size_t ws_size, hipStream_t stream) {
  const int*   ids  = (const int*)d_in[0];
  const float* mask = (const float*)d_in[1];
  const float* emb  = (const float*)d_in[2];
  const float* x0   = (const float*)d_in[3];
  const float* v0   = (const float*)d_in[4];
  const float* U    = (const float*)d_in[5];
  const float* Wc   = (const float*)d_in[6];
  const float* G    = (const float*)d_in[7];
  const float* Cw   = (const float*)d_in[8];
  const float* Cb   = (const float*)d_in[9];
  const float* lnw  = (const float*)d_in[10];
  const float* lnb  = (const float*)d_in[11];
  const float* row  = (const float*)d_in[12];
  const float* rob  = (const float*)d_in[13];
  float* out = (float*)d_out;

  // workspace: A-matrix f16 [8192x512], then packed ro_w u32 [D_/2][WP_]
  _Float16* Af16 = (_Float16*)d_ws;
  unsigned* Whp =
      (unsigned*)((char*)d_ws + (size_t)B_ * T_ * D_ * sizeof(_Float16));

  const int nW = (D_ / 2) * WP_;
  cvt_pad_pack<<<nW / 256, 256, 0, stream>>>(row, Whp);

  float* tail = out + (size_t)B_ * T_ * V_;
  scan_kernel<<<B_, 256, 0, stream>>>(ids, mask, emb, x0, v0, U, Wc, G, Cw, Cb,
                                      lnw, lnb, Af16, tail);

  dim3 grid((B_ * T_) / BM, WP_ / BN);
  readout_gemm<<<grid, dim3(256), 0, stream>>>(Af16, Whp, rob, out);
}
